// BottleNeck_69810398429628
// MI455X (gfx1250) — matrix operand order repaired and checked by reading, not compiled, not run on hardware
//
#include <hip/hip_runtime.h>

typedef __attribute__((ext_vector_type(16))) __bf16 v16bf;
typedef __attribute__((ext_vector_type(8)))  __bf16 v8bf;
typedef __attribute__((ext_vector_type(8)))  float  v8f;

#define TB   256
#define BMT  64
#define BNT  64
#define BKT  64
#define BKP  72   // padded LDS pitch (144B, 16B-aligned, avoids 256B bank period)
#define KNNK 10
#define NPTS 4096
#define FD   256

static __device__ __forceinline__ __bf16 f2bf(float v) { return (__bf16)v; }

static __device__ __forceinline__ v16bf cat16(v8bf lo, v8bf hi) {
  return __builtin_shufflevector(lo, hi, 0, 1, 2, 3, 4, 5, 6, 7,
                                 8, 9, 10, 11, 12, 13, 14, 15);
}

// ---------------------------------------------------------------------------
// Generic bf16-WMMA GEMM:  C[M,N] = alpha * A(MxK) @ B(KxN) (+bias) (+addsrc)
// BMODE 0: B dense, B[k][n] = Bp[k*ldb + n]
// BMODE 1: B = graph feature of X (cfeat x NPTS) via nbr idx; N = NPTS*KNNK
// a_trans: A[i][k] = Ap[k*lda + i]   (else Ap[i*lda + k])
// Optional per-row sum / sum-of-squares atomics for instance-norm stats.
// LDS: A staged [m][k] (pitch BKP), B staged transposed [n][k] (pitch BKP) so
// every lane's WMMA fragment is contiguous -> ds_load_b128 pairs, no packing.
// ---------------------------------------------------------------------------
template <int BMODE>
__global__ __launch_bounds__(TB) void gemm_bf16_k(
    const float* __restrict__ Ap, long lda, int a_trans,
    const float* __restrict__ Bp, long ldb,
    const float* __restrict__ X, const int* __restrict__ nidx, int cfeat,
    const float* __restrict__ bias, const float* __restrict__ addsrc,
    float* __restrict__ Cp, long ldc,
    int M, int N, int K, float alpha,
    float* __restrict__ ssum, float* __restrict__ ssq)
{
  __shared__ __attribute__((aligned(16))) __bf16 sA[BMT * BKP];
  __shared__ __attribute__((aligned(16))) __bf16 sB[BNT * BKP];

  const int tid  = threadIdx.x;
  const int lane = tid & 31;
  const int wave = tid >> 5;
  const int n0 = blockIdx.x * BNT;
  const int m0 = blockIdx.y * BMT;

  const int mt  = wave >> 1;           // 0..3 (16-row subtile)
  const int nt0 = (wave * 2) & 3;      // two adjacent 16-col subtiles
  const int nt1 = nt0 + 1;

  v8f acc0 = {};
  v8f acc1 = {};

  const int hf = lane >> 4;            // wave half (0/1)
  const int ml = lane & 15;

  for (int kb = 0; kb < K; kb += BKT) {
    // ---- stage A tile (BMT x BKT) as bf16, pitch BKP ----
    for (int e = tid; e < BMT * BKT; e += TB) {
      int r  = e >> 6;
      int kk = e & 63;
      int gi = m0 + r, gk = kb + kk;
      float v = 0.f;
      if (gi < M && gk < K)
        v = a_trans ? Ap[(long)gk * lda + gi] : Ap[(long)gi * lda + gk];
      sA[r * BKP + kk] = f2bf(v);
    }
    // ---- stage B tile transposed ([n][k]) as bf16, pitch BKP ----
    for (int e = tid; e < BKT * BNT; e += TB) {
      int kk = e >> 6;
      int c  = e & 63;
      int gk = kb + kk, gn = n0 + c;
      float v = 0.f;
      if (gk < K && gn < N) {
        if (BMODE == 0) {
          v = Bp[(long)gk * ldb + gn];
        } else {
          int pt = gn / KNNK;
          int j  = gn - pt * KNNK;
          if (gk < cfeat) {
            v = X[(long)gk * NPTS + pt];
          } else {
            int kc = gk - cfeat;
            int nb = nidx[pt * KNNK + j];
            v = X[(long)kc * NPTS + nb] - X[(long)kc * NPTS + pt];
          }
        }
      }
      sB[c * BKP + kk] = f2bf(v);
    }
    __syncthreads();

    // CDNA5 global prefetch of next B tile (speculative, no counter cost)
    if (BMODE == 0 && (kb + BKT) < K && (n0 + BNT) <= N) {
      const float* pf = Bp + (long)(kb + BKT + (tid >> 2)) * ldb + n0 + (tid & 3) * 16;
      __builtin_prefetch(pf, 0, 1);
    }

    #pragma unroll
    for (int ks = 0; ks < BKT; ks += 32) {
      // A fragment: k = ks + hf*16 + e -> one contiguous 32B run per lane
      const __bf16* pa = &sA[(mt * 16 + ml) * BKP + ks + (hf << 4)];
      v16bf af = cat16(*(const v8bf*)pa, *(const v8bf*)(pa + 8));
      // B fragment: k = ks + hf*16 + e -> one contiguous 32B run per lane
      const __bf16* pb0 = &sB[(nt0 * 16 + ml) * BKP + ks + (hf << 4)];
      v16bf bf0 = cat16(*(const v8bf*)pb0, *(const v8bf*)(pb0 + 8));
      const __bf16* pb1 = &sB[(nt1 * 16 + ml) * BKP + ks + (hf << 4)];
      v16bf bf1 = cat16(*(const v8bf*)pb1, *(const v8bf*)(pb1 + 8));

      acc0 = __builtin_amdgcn_wmma_f32_16x16x32_bf16(false, af, false, bf0,
                                                     (short)0, acc0, false, false);
      acc1 = __builtin_amdgcn_wmma_f32_16x16x32_bf16(false, af, false, bf1,
                                                     (short)0, acc1, false, false);
    }
    __syncthreads();
  }

  // ---- epilogue: C layout reg j -> row (j + 8*half), col = lane%16 ----
  #pragma unroll
  for (int j = 0; j < 8; j++) {
    int row = m0 + mt * 16 + j + (hf << 3);
    if (row < M) {
      float bi = bias ? bias[row] : 0.f;
      {
        int col = n0 + nt0 * 16 + ml;
        if (col < N) {
          float v = alpha * acc0[j] + bi;
          if (addsrc) v += addsrc[(long)row * ldc + col];
          Cp[(long)row * ldc + col] = v;
          if (ssum) { atomicAdd(&ssum[row], v); atomicAdd(&ssq[row], v * v); }
        }
      }
      {
        int col = n0 + nt1 * 16 + ml;
        if (col < N) {
          float v = alpha * acc1[j] + bi;
          if (addsrc) v += addsrc[(long)row * ldc + col];
          Cp[(long)row * ldc + col] = v;
          if (ssum) { atomicAdd(&ssum[row], v); atomicAdd(&ssq[row], v * v); }
        }
      }
    }
  }
}

// ---------------------------------------------------------------------------
// KNN: per query row, top-10 smallest squared distances (includes self).
// ---------------------------------------------------------------------------
__global__ __launch_bounds__(TB) void knn_kernel(const float* __restrict__ pts,
                                                 int* __restrict__ outidx, int n)
{
  int row = blockIdx.x;
  float px = pts[row], py = pts[n + row], pz = pts[2 * n + row];
  float best[KNNK]; int bidx[KNNK];
  #pragma unroll
  for (int i = 0; i < KNNK; i++) { best[i] = 3.4e38f; bidx[i] = 0; }
  for (int m = threadIdx.x; m < n; m += TB) {
    float dx = px - pts[m], dy = py - pts[n + m], dz = pz - pts[2 * n + m];
    float d = dx * dx + dy * dy + dz * dz;
    if (d < best[KNNK - 1]) {
      int p = KNNK - 1;
      while (p > 0 && best[p - 1] > d) { best[p] = best[p - 1]; bidx[p] = bidx[p - 1]; p--; }
      best[p] = d; bidx[p] = m;
    }
  }
  __shared__ float sd[TB * KNNK];
  __shared__ int   si[TB * KNNK];
  __shared__ float rd[TB];
  __shared__ int   ri[TB];
  for (int i = 0; i < KNNK; i++) {
    sd[threadIdx.x * KNNK + i] = best[i];
    si[threadIdx.x * KNNK + i] = bidx[i];
  }
  __syncthreads();
  for (int r = 0; r < KNNK; r++) {
    float md = 3.4e38f; int ms = 0;
    for (int i = 0; i < KNNK; i++) {
      float d = sd[threadIdx.x * KNNK + i];
      if (d < md) { md = d; ms = threadIdx.x * KNNK + i; }
    }
    rd[threadIdx.x] = md; ri[threadIdx.x] = ms;
    __syncthreads();
    for (int s = TB / 2; s > 0; s >>= 1) {
      if (threadIdx.x < s && rd[threadIdx.x + s] < rd[threadIdx.x]) {
        rd[threadIdx.x] = rd[threadIdx.x + s]; ri[threadIdx.x] = ri[threadIdx.x + s];
      }
      __syncthreads();
    }
    if (threadIdx.x == 0) {
      int slot = ri[0];
      outidx[row * KNNK + r] = si[slot];
      sd[slot] = 3.4e38f;
    }
    __syncthreads();
  }
}

// ---------------------------------------------------------------------------
// Small helpers
// ---------------------------------------------------------------------------
__global__ void copy_f32(const float* __restrict__ s, float* __restrict__ d, int n)
{ int i = blockIdx.x * TB + threadIdx.x; if (i < n) d[i] = s[i]; }

__global__ void finalize_stats(const float* ssum, const float* ssq,
                               float* mean, float* rstd, int C, float invcnt)
{
  int i = blockIdx.x * TB + threadIdx.x;
  if (i < C) {
    float m = ssum[i] * invcnt;
    float v = ssq[i] * invcnt - m * m;
    mean[i] = m;
    rstd[i] = rsqrtf(v + 1e-5f);
  }
}

__global__ void inorm_lrelu_maxpool(const float* __restrict__ E, const float* mean,
                                    const float* rstd, float* __restrict__ out,
                                    int C, int npts)
{
  int t = blockIdx.x * TB + threadIdx.x;
  if (t >= C * npts) return;
  int c = t / npts, nn = t - c * npts;
  float m = mean[c], r = rstd[c];
  const float* e = E + (long)c * npts * KNNK + (long)nn * KNNK;
  float mx = -3.4e38f;
  #pragma unroll
  for (int j = 0; j < KNNK; j++) {
    float v = (e[j] - m) * r;
    v = v >= 0.f ? v : 0.2f * v;
    mx = fmaxf(mx, v);
  }
  out[(long)c * npts + nn] = mx;
}

__global__ void inorm_act(const float* __restrict__ Y, const float* mean,
                          const float* rstd, float* __restrict__ out,
                          int C, int npts, int act)  // 1=lrelu 2=relu
{
  int t = blockIdx.x * TB + threadIdx.x;
  if (t >= C * npts) return;
  int c = t / npts, nn = t - c * npts;
  float v = (Y[(long)c * npts + nn] - mean[c]) * rstd[c];
  if (act == 1) v = v >= 0.f ? v : 0.2f * v;
  else if (act == 2) v = fmaxf(v, 0.f);
  out[(long)c * npts + nn] = v;
}

// softmax over rows (m) for each column n: S is rows x cols, in place
__global__ __launch_bounds__(TB) void softmax_col(float* __restrict__ S, int rows, int cols)
{
  int n = blockIdx.x;
  __shared__ float red[TB];
  float mx = -3.4e38f;
  for (int m = threadIdx.x; m < rows; m += TB) mx = fmaxf(mx, S[(long)m * cols + n]);
  red[threadIdx.x] = mx; __syncthreads();
  for (int s = TB / 2; s > 0; s >>= 1) {
    if (threadIdx.x < s) red[threadIdx.x] = fmaxf(red[threadIdx.x], red[threadIdx.x + s]);
    __syncthreads();
  }
  mx = red[0]; __syncthreads();
  float sum = 0.f;
  for (int m = threadIdx.x; m < rows; m += TB) {
    float e = __expf(S[(long)m * cols + n] - mx);
    S[(long)m * cols + n] = e; sum += e;
  }
  red[threadIdx.x] = sum; __syncthreads();
  for (int s = TB / 2; s > 0; s >>= 1) {
    if (threadIdx.x < s) red[threadIdx.x] += red[threadIdx.x + s];
    __syncthreads();
  }
  float inv = 1.f / red[0];
  for (int m = threadIdx.x; m < rows; m += TB) S[(long)m * cols + n] *= inv;
}

__global__ void l2norm_cols(const float* __restrict__ F, float* __restrict__ Fn,
                            int C, int cols)
{
  int j = blockIdx.x * TB + threadIdx.x;
  if (j >= cols) return;
  float s = 0.f;
  for (int c = 0; c < C; c++) { float v = F[(long)c * cols + j]; s += v * v; }
  float inv = 1.f / fmaxf(sqrtf(s), 1e-12f);
  for (int c = 0; c < C; c++) Fn[(long)c * cols + j] = F[(long)c * cols + j] * inv;
}

// sal[i] = softmax_j(ip[i][j]/temp) . sr[j]
__global__ __launch_bounds__(TB) void sal_row(const float* __restrict__ ip,
                                              const float* __restrict__ sr,
                                              const float* __restrict__ eps,
                                              float* __restrict__ sal, int n)
{
  int i = blockIdx.x;
  float invt = 1.f / (__expf(eps[0]) + 0.03f);
  __shared__ float r1[TB]; __shared__ float r2[TB];
  float mx = -3.4e38f;
  for (int j = threadIdx.x; j < n; j += TB) mx = fmaxf(mx, ip[(long)i * n + j] * invt);
  r1[threadIdx.x] = mx; __syncthreads();
  for (int s = TB / 2; s > 0; s >>= 1) {
    if (threadIdx.x < s) r1[threadIdx.x] = fmaxf(r1[threadIdx.x], r1[threadIdx.x + s]);
    __syncthreads();
  }
  mx = r1[0]; __syncthreads();
  float se = 0.f, sd = 0.f;
  for (int j = threadIdx.x; j < n; j += TB) {
    float e = __expf(ip[(long)i * n + j] * invt - mx);
    se += e; sd += e * sr[j];
  }
  r1[threadIdx.x] = se; r2[threadIdx.x] = sd; __syncthreads();
  for (int s = TB / 2; s > 0; s >>= 1) {
    if (threadIdx.x < s) { r1[threadIdx.x] += r1[threadIdx.x + s]; r2[threadIdx.x] += r2[threadIdx.x + s]; }
    __syncthreads();
  }
  if (threadIdx.x == 0) sal[i] = r2[0] / r1[0];
}

// sal[j] = softmax_i(ip[i][j]/temp) . sr[i]
__global__ __launch_bounds__(TB) void sal_col(const float* __restrict__ ip,
                                              const float* __restrict__ sr,
                                              const float* __restrict__ eps,
                                              float* __restrict__ sal, int n)
{
  int j = blockIdx.x;
  float invt = 1.f / (__expf(eps[0]) + 0.03f);
  __shared__ float r1[TB]; __shared__ float r2[TB];
  float mx = -3.4e38f;
  for (int i = threadIdx.x; i < n; i += TB) mx = fmaxf(mx, ip[(long)i * n + j] * invt);
  r1[threadIdx.x] = mx; __syncthreads();
  for (int s = TB / 2; s > 0; s >>= 1) {
    if (threadIdx.x < s) r1[threadIdx.x] = fmaxf(r1[threadIdx.x], r1[threadIdx.x + s]);
    __syncthreads();
  }
  mx = r1[0]; __syncthreads();
  float se = 0.f, sd = 0.f;
  for (int i = threadIdx.x; i < n; i += TB) {
    float e = __expf(ip[(long)i * n + j] * invt - mx);
    se += e; sd += e * sr[i];
  }
  r1[threadIdx.x] = se; r2[threadIdx.x] = sd; __syncthreads();
  for (int s = TB / 2; s > 0; s >>= 1) {
    if (threadIdx.x < s) { r1[threadIdx.x] += r1[threadIdx.x + s]; r2[threadIdx.x] += r2[threadIdx.x + s]; }
    __syncthreads();
  }
  if (threadIdx.x == 0) sal[j] = r2[0] / r1[0];
}

// out (rows x 258): [sraw, sal, fraw(256)]
__global__ void assemble_out(const float* __restrict__ sraw, const float* __restrict__ sal,
                             const float* __restrict__ fc, float* __restrict__ out,
                             int rows, int C)
{
  int i = blockIdx.x;
  if (threadIdx.x == 0) { out[(long)i * 258 + 0] = sraw[i]; out[(long)i * 258 + 1] = sal[i]; }
  for (int c = threadIdx.x; c < C; c += blockDim.x)
    out[(long)i * 258 + 2 + c] = fc[(long)c * rows + i];
}

// ---------------------------------------------------------------------------
// Host-side orchestration helpers
// ---------------------------------------------------------------------------
static void gemm(hipStream_t s, int mode,
                 const float* A, long lda, int atr,
                 const float* B, long ldb,
                 const float* X, const int* nidx, int cfeat,
                 const float* bias, const float* addsrc,
                 float* C, long ldc, int M, int N, int K, float alpha,
                 float* ssum, float* ssq)
{
  dim3 g((N + BNT - 1) / BNT, (M + BMT - 1) / BMT);
  if (mode == 0)
    gemm_bf16_k<0><<<g, TB, 0, s>>>(A, lda, atr, B, ldb, X, nidx, cfeat,
                                    bias, addsrc, C, ldc, M, N, K, alpha, ssum, ssq);
  else
    gemm_bf16_k<1><<<g, TB, 0, s>>>(A, lda, atr, B, ldb, X, nidx, cfeat,
                                    bias, addsrc, C, ldc, M, N, K, alpha, ssum, ssq);
}

static void run_gnn(hipStream_t s, const int* nidx, const float* fin,
                    const float* w1, const float* w2, const float* w3,
                    float* X3, float* E, float* outF,
                    float* ssum, float* ssq, float* mean, float* rstd)
{
  const long NK = (long)NPTS * KNNK;
  copy_f32<<<(FD * NPTS + TB - 1) / TB, TB, 0, s>>>(fin, X3, FD * NPTS);

  hipMemsetAsync(ssum, 0, 1024 * 4, s); hipMemsetAsync(ssq, 0, 1024 * 4, s);
  gemm(s, 1, w1, 512, 0, nullptr, 0, X3, nidx, 256, nullptr, nullptr,
       E, NK, 256, (int)NK, 512, 1.f, ssum, ssq);
  finalize_stats<<<4, TB, 0, s>>>(ssum, ssq, mean, rstd, 256, 1.f / (float)NK);
  inorm_lrelu_maxpool<<<(256 * NPTS + TB - 1) / TB, TB, 0, s>>>(E, mean, rstd,
                                                                X3 + 256 * NPTS, 256, NPTS);

  hipMemsetAsync(ssum, 0, 1024 * 4, s); hipMemsetAsync(ssq, 0, 1024 * 4, s);
  gemm(s, 1, w2, 512, 0, nullptr, 0, X3 + 256 * NPTS, nidx, 256, nullptr, nullptr,
       E, NK, 512, (int)NK, 512, 1.f, ssum, ssq);
  finalize_stats<<<4, TB, 0, s>>>(ssum, ssq, mean, rstd, 512, 1.f / (float)NK);
  inorm_lrelu_maxpool<<<(512 * NPTS + TB - 1) / TB, TB, 0, s>>>(E, mean, rstd,
                                                                X3 + 512 * NPTS, 512, NPTS);

  hipMemsetAsync(ssum, 0, 1024 * 4, s); hipMemsetAsync(ssq, 0, 1024 * 4, s);
  gemm(s, 0, w3, 1024, 0, X3, NPTS, nullptr, nullptr, 0, nullptr, nullptr,
       E, NPTS, 256, NPTS, 1024, 1.f, ssum, ssq);
  finalize_stats<<<4, TB, 0, s>>>(ssum, ssq, mean, rstd, 256, 1.f / (float)NPTS);
  inorm_act<<<(256 * NPTS + TB - 1) / TB, TB, 0, s>>>(E, mean, rstd, outF, 256, NPTS, 1);
}

static void run_cross(hipStream_t s, const float* srcF, const float* tgtF, float* outF,
                      const float* qw, const float* qb, const float* kw, const float* kb,
                      const float* vw, const float* vb, const float* mw, const float* mb,
                      const float* m1w, const float* m1b, const float* m2w, const float* m2b,
                      float* Qb_, float* Kb_, float* Vb_, float* Msg, float* Hbuf, float* Sbuf,
                      float* ssum, float* ssq, float* mean, float* rstd)
{
  gemm(s, 0, qw, 256, 0, srcF, NPTS, nullptr, nullptr, 0, qb, nullptr,
       Qb_, NPTS, 256, NPTS, 256, 1.f, nullptr, nullptr);
  gemm(s, 0, kw, 256, 0, tgtF, NPTS, nullptr, nullptr, 0, kb, nullptr,
       Kb_, NPTS, 256, NPTS, 256, 1.f, nullptr, nullptr);
  gemm(s, 0, vw, 256, 0, tgtF, NPTS, nullptr, nullptr, 0, vb, nullptr,
       Vb_, NPTS, 256, NPTS, 256, 1.f, nullptr, nullptr);

  for (int h = 0; h < 4; h++) {
    // S[m,n] = (1/8) * sum_d K[d,h,m] Q[d,h,n]   (channel c = d*4 + h)
    gemm(s, 0, Kb_ + h * NPTS, 4L * NPTS, 1, Qb_ + h * NPTS, 4L * NPTS,
         nullptr, nullptr, 0, nullptr, nullptr,
         Sbuf, NPTS, NPTS, NPTS, 64, 0.125f, nullptr, nullptr);
    softmax_col<<<NPTS, TB, 0, s>>>(Sbuf, NPTS, NPTS);   // softmax over m per column n
    // msg[d,n] = sum_m V[d,h,m] P[m,n]
    gemm(s, 0, Vb_ + h * NPTS, 4L * NPTS, 0, Sbuf, NPTS,
         nullptr, nullptr, 0, nullptr, nullptr,
         Msg + h * NPTS, 4L * NPTS, 64, NPTS, NPTS, 1.f, nullptr, nullptr);
  }

  gemm(s, 0, mw, 256, 0, Msg, NPTS, nullptr, nullptr, 0, mb, nullptr,
       Hbuf + 256 * NPTS, NPTS, 256, NPTS, 256, 1.f, nullptr, nullptr);
  copy_f32<<<(256 * NPTS + TB - 1) / TB, TB, 0, s>>>(srcF, Hbuf, 256 * NPTS);

  hipMemsetAsync(ssum, 0, 1024 * 4, s); hipMemsetAsync(ssq, 0, 1024 * 4, s);
  gemm(s, 0, m1w, 512, 0, Hbuf, NPTS, nullptr, nullptr, 0, m1b, nullptr,
       Sbuf, NPTS, 512, NPTS, 512, 1.f, ssum, ssq);
  finalize_stats<<<4, TB, 0, s>>>(ssum, ssq, mean, rstd, 512, 1.f / (float)NPTS);
  inorm_act<<<(512 * NPTS + TB - 1) / TB, TB, 0, s>>>(Sbuf, mean, rstd, Hbuf, 512, NPTS, 2);

  gemm(s, 0, m2w, 512, 0, Hbuf, NPTS, nullptr, nullptr, 0, m2b, srcF,
       outF, NPTS, 256, NPTS, 512, 1.f, nullptr, nullptr);
}

// ---------------------------------------------------------------------------
extern "C" void kernel_launch(void* const* d_in, const int* in_sizes, int n_in,
                              void* d_out, int out_size, void* d_ws, size_t ws_size,
                              hipStream_t stream)
{
  const float* points_src = (const float*)d_in[0];
  const float* feats_src  = (const float*)d_in[1];
  const float* points_tgt = (const float*)d_in[2];
  const float* feats_tgt  = (const float*)d_in[3];
  const float* g1_w1 = (const float*)d_in[4];
  const float* g1_w2 = (const float*)d_in[5];
  const float* g1_w3 = (const float*)d_in[6];
  const float* g2_w1 = (const float*)d_in[7];
  const float* g2_w2 = (const float*)d_in[8];
  const float* g2_w3 = (const float*)d_in[9];
  const float* ca_qw = (const float*)d_in[10];
  const float* ca_qb = (const float*)d_in[11];
  const float* ca_kw = (const float*)d_in[12];
  const float* ca_kb = (const float*)d_in[13];
  const float* ca_vw = (const float*)d_in[14];
  const float* ca_vb = (const float*)d_in[15];
  const float* ca_mw = (const float*)d_in[16];
  const float* ca_mb = (const float*)d_in[17];
  const float* ca_m1w = (const float*)d_in[18];
  const float* ca_m1b = (const float*)d_in[19];
  const float* ca_m2w = (const float*)d_in[20];
  const float* ca_m2b = (const float*)d_in[21];
  const float* fp_w = (const float*)d_in[22];
  const float* fp_b = (const float*)d_in[23];
  const float* sp_w = (const float*)d_in[24];
  const float* sp_b = (const float*)d_in[25];
  const float* eps  = (const float*)d_in[26];

  // ---- workspace bump allocator ----
  char* wp = (char*)d_ws;
  auto alloc = [&](size_t b) -> char* {
    char* p = wp; wp += (b + 255) & ~(size_t)255; return p;
  };
  int*   idxS = (int*)alloc((size_t)NPTS * KNNK * 4);
  int*   idxT = (int*)alloc((size_t)NPTS * KNNK * 4);
  float* BIG  = (float*)alloc((size_t)512 * NPTS * KNNK * 4);   // E / S / m1-temp union
  float* X3   = (float*)alloc((size_t)1024 * NPTS * 4);
  float* sfA  = (float*)alloc((size_t)FD * NPTS * 4);
  float* tfA  = (float*)alloc((size_t)FD * NPTS * 4);
  float* sfB  = (float*)alloc((size_t)FD * NPTS * 4);
  float* tfB  = (float*)alloc((size_t)FD * NPTS * 4);
  float* sfC  = (float*)alloc((size_t)FD * NPTS * 4);
  float* tfC  = (float*)alloc((size_t)FD * NPTS * 4);
  float* Qb_  = (float*)alloc((size_t)FD * NPTS * 4);
  float* Kb_  = (float*)alloc((size_t)FD * NPTS * 4);
  float* Vb_  = (float*)alloc((size_t)FD * NPTS * 4);
  float* Msg  = (float*)alloc((size_t)FD * NPTS * 4);
  float* Hbuf = (float*)alloc((size_t)512 * NPTS * 4);
  float* FC   = (float*)alloc((size_t)FD * 2 * NPTS * 4);
  float* FN   = (float*)alloc((size_t)FD * 2 * NPTS * 4);
  float* SRAW = (float*)alloc((size_t)2 * NPTS * 4);
  float* SAL  = (float*)alloc((size_t)2 * NPTS * 4);
  float* ssum = (float*)alloc(1024 * 4);
  float* ssq  = (float*)alloc(1024 * 4);
  float* mean = (float*)alloc(1024 * 4);
  float* rstd = (float*)alloc(1024 * 4);

  // ---- KNN (once per point set; reused by g1 and g2) ----
  knn_kernel<<<NPTS, TB, 0, stream>>>(points_src, idxS, NPTS);
  knn_kernel<<<NPTS, TB, 0, stream>>>(points_tgt, idxT, NPTS);

  // ---- GNN stage 1 ----
  run_gnn(stream, idxS, feats_src, g1_w1, g1_w2, g1_w3, X3, BIG, sfA, ssum, ssq, mean, rstd);
  run_gnn(stream, idxT, feats_tgt, g1_w1, g1_w2, g1_w3, X3, BIG, tfA, ssum, ssq, mean, rstd);

  // ---- cross attention updates ----
  run_cross(stream, sfA, tfA, sfB, ca_qw, ca_qb, ca_kw, ca_kb, ca_vw, ca_vb,
            ca_mw, ca_mb, ca_m1w, ca_m1b, ca_m2w, ca_m2b,
            Qb_, Kb_, Vb_, Msg, Hbuf, BIG, ssum, ssq, mean, rstd);
  run_cross(stream, tfA, sfB, tfB, ca_qw, ca_qb, ca_kw, ca_kb, ca_vw, ca_vb,
            ca_mw, ca_mb, ca_m1w, ca_m1b, ca_m2w, ca_m2b,
            Qb_, Kb_, Vb_, Msg, Hbuf, BIG, ssum, ssq, mean, rstd);

  // ---- GNN stage 2 ----
  run_gnn(stream, idxS, sfB, g2_w1, g2_w2, g2_w3, X3, BIG, sfC, ssum, ssq, mean, rstd);
  run_gnn(stream, idxT, tfB, g2_w1, g2_w2, g2_w3, X3, BIG, tfC, ssum, ssq, mean, rstd);

  // ---- final projections ----
  gemm(stream, 0, fp_w, 256, 0, sfC, NPTS, nullptr, nullptr, 0, fp_b, nullptr,
       FC, 2 * NPTS, 256, NPTS, 256, 1.f, nullptr, nullptr);
  gemm(stream, 0, fp_w, 256, 0, tfC, NPTS, nullptr, nullptr, 0, fp_b, nullptr,
       FC + NPTS, 2 * NPTS, 256, NPTS, 256, 1.f, nullptr, nullptr);
  gemm(stream, 0, sp_w, 256, 0, FC, 2 * NPTS, nullptr, nullptr, 0, sp_b, nullptr,
       SRAW, 2 * NPTS, 1, 2 * NPTS, 256, 1.f, nullptr, nullptr);

  l2norm_cols<<<(2 * NPTS + TB - 1) / TB, TB, 0, stream>>>(FC, FN, 256, 2 * NPTS);

  // ip = fg[:N] @ fg[N:].T   (A = FN^T left half, B = FN right half)
  gemm(stream, 0, FN, 2L * NPTS, 1, FN + NPTS, 2L * NPTS, nullptr, nullptr, 0,
       nullptr, nullptr, BIG, NPTS, NPTS, NPTS, 256, 1.f, nullptr, nullptr);

  sal_row<<<NPTS, TB, 0, stream>>>(BIG, SRAW + NPTS, eps, SAL, NPTS);
  sal_col<<<NPTS, TB, 0, stream>>>(BIG, SRAW, eps, SAL + NPTS, NPTS);

  assemble_out<<<2 * NPTS, TB, 0, stream>>>(SRAW, SAL, FC, (float*)d_out, 2 * NPTS, 256);
}
